// EncoderDecoderWrapper_13924283974438
// MI455X (gfx1250) — matrix operand order repaired and checked by reading, not compiled, not run on hardware
//
#include <hip/hip_runtime.h>

#define B_ 512
#define S_ 336
#define F_ 16
#define H_ 256
#define P_ 48
#define NATT 384  // attention N padded to 24 WMMA tiles (6 per wave, no guards)

typedef __bf16 bf16_t;
typedef __attribute__((ext_vector_type(8)))  __bf16 v8bf;
typedef __attribute__((ext_vector_type(16))) __bf16 v16bf;
typedef __attribute__((ext_vector_type(8)))  float  v8f;

static __device__ __forceinline__ v8bf ldg8(const bf16_t* p) {
  return *(const v8bf*)p;
}
static __device__ __forceinline__ v16bf cat16(v8bf lo, v8bf hi) {
  v16bf r;
#pragma unroll
  for (int i = 0; i < 8; ++i) { r[i] = lo[i]; r[i + 8] = hi[i]; }
  return r;
}

// ---- fused GRU step: WMMA gate GEMMs + LDS gate recombination + h update ----
// A = [h (256) | x (XS)] row-major; B = pre-packed gate-major Bm[4*H][KTOT]
//   (Bm rows g*H+col hold [Whh_g | Wih_g | 0] with nh/nx zero regions baked in)
// grid (8,8): blockIdx.x -> 64 batch rows, blockIdx.y -> 32 hidden cols
// 8 waves: wm = row half, wg = gate group {r, z, nh, nx}
template <int XS, int KTOT, bool SEQ>
__global__ __launch_bounds__(256) void gru_step_kernel(
    const bf16_t* __restrict__ h_bf, const float* __restrict__ h_f32,
    const bf16_t* __restrict__ x_bf, const bf16_t* __restrict__ Bm,
    const float* __restrict__ bih, const float* __restrict__ bhh,
    float* __restrict__ h_out_f32, bf16_t* __restrict__ h_out_bf,
    bf16_t* __restrict__ seq_out) {
  __shared__ float G[4][64][33];
  const int tid  = threadIdx.x;
  const int lane = tid & 31;
  const int wid  = tid >> 5;
  const int wm   = wid >> 2;
  const int wg   = wid & 3;
  const int m0   = blockIdx.x * 64;
  const int j0   = blockIdx.y * 32;
  const int rsel = lane & 15;
  const int kbA  = (lane < 16) ? 0 : 8;   // A: two 8-half chunks at +0/+16
  const int kbB  = (lane < 16) ? 0 : 8;   // B: two 8-half chunks at +0/+16

  // hoisted per-lane row bases (WMMA operand layouts per CDNA5 ISA 7.12.2)
  const bf16_t* hrow[2];
  const bf16_t* xrow[2];
  const bf16_t* brow[2];
#pragma unroll
  for (int i = 0; i < 2; ++i) {
    const int row = m0 + wm * 32 + i * 16 + rsel;
    hrow[i] = h_bf + (size_t)row * H_ + kbA;
    xrow[i] = x_bf + (size_t)row * XS + kbA;
  }
#pragma unroll
  for (int j = 0; j < 2; ++j) {
    const int col = j0 + j * 16 + rsel;
    brow[j] = Bm + (size_t)(wg * H_ + col) * KTOT + kbB;
  }

  v8f acc[2][2];
#pragma unroll
  for (int i = 0; i < 2; ++i)
#pragma unroll
    for (int j = 0; j < 2; ++j)
#pragma unroll
      for (int e = 0; e < 8; ++e) acc[i][j][e] = 0.0f;

#pragma unroll
  for (int kc = 0; kc < KTOT; kc += 32) {
    v16bf af[2], bfr[2];
#pragma unroll
    for (int i = 0; i < 2; ++i) {
      // region select folds at compile time (kc is an unrolled constant;
      // chunk bases are 16-aligned, region boundary 256 is 32-aligned)
      v8bf lo = (kc < H_)      ? ldg8(hrow[i] + kc)
                               : ldg8(xrow[i] + (kc - H_));
      v8bf hi = (kc + 16 < H_) ? ldg8(hrow[i] + kc + 16)
                               : ldg8(xrow[i] + (kc + 16 - H_));
      af[i] = cat16(lo, hi);
    }
#pragma unroll
    for (int j = 0; j < 2; ++j)
      bfr[j] = cat16(ldg8(brow[j] + kc), ldg8(brow[j] + kc + 16));
#pragma unroll
    for (int i = 0; i < 2; ++i)
#pragma unroll
      for (int j = 0; j < 2; ++j)
        acc[i][j] = __builtin_amdgcn_wmma_f32_16x16x32_bf16(
            false, af[i], false, bfr[j], (short)0, acc[i][j], false, false);
  }

  // spill accumulators: C/D layout -> VGPR v holds M=v (lanes 0-15) / M=v+8 (16-31)
#pragma unroll
  for (int i = 0; i < 2; ++i)
#pragma unroll
    for (int j = 0; j < 2; ++j)
#pragma unroll
      for (int v = 0; v < 8; ++v) {
        const int rl = wm * 32 + i * 16 + v + ((lane >> 4) << 3);
        const int cl = j * 16 + rsel;
        G[wg][rl][cl] = acc[i][j][v];
      }
  __syncthreads();

#pragma unroll
  for (int e = 0; e < 8; ++e) {
    const int idx = tid + e * 256;
    const int rl  = idx >> 5;
    const int jj  = idx & 31;
    const int b   = m0 + rl;
    const int j   = j0 + jj;
    const float rp = G[0][rl][jj];
    const float zp = G[1][rl][jj];
    const float nh = G[2][rl][jj];
    const float nx = G[3][rl][jj];
    const float r  = 1.0f / (1.0f + expf(-(rp + bih[j] + bhh[j])));
    const float z  = 1.0f / (1.0f + expf(-(zp + bih[H_ + j] + bhh[H_ + j])));
    const float n  = tanhf(nx + bih[2 * H_ + j] + r * (nh + bhh[2 * H_ + j]));
    const float hv = h_f32[(size_t)b * H_ + j];
    const float ho = (1.0f - z) * n + z * hv;
    h_out_f32[(size_t)b * H_ + j] = ho;
    const bf16_t hob = (bf16_t)ho;
    h_out_bf[(size_t)b * H_ + j] = hob;
    if (SEQ) seq_out[(size_t)b * S_ * H_ + j] = hob;
  }
}

// ---- attention scores + softmax (fused), 16 blocks of 32 rows ----
// A = [h (256) | y_pad (32)]; B = attB[NATT][288] (att_W padded 272->288,
// N padded 336->384 with zero rows so every wave runs 6 unguarded tiles)
__global__ __launch_bounds__(256) void att_kernel(
    const bf16_t* __restrict__ h_bf, const bf16_t* __restrict__ y_pad,
    const bf16_t* __restrict__ attB, const float* __restrict__ att_bp,
    float* __restrict__ attn) {
  __shared__ float SC[32][NATT + 1];
  __shared__ float SM[32], SR[32];
  const int tid  = threadIdx.x;
  const int lane = tid & 31;
  const int wid  = tid >> 5;
  const int wm   = wid >> 2;
  const int wn   = wid & 3;
  const int m0   = blockIdx.x * 32;
  const int rsel = lane & 15;
  const int kbA  = (lane < 16) ? 0 : 8;
  const int kbB  = (lane < 16) ? 0 : 8;
  const int nt0  = wn * 6;  // 24 N-tiles, 6 per wave, unconditional

  const int row = m0 + wm * 16 + rsel;
  const bf16_t* hrow = h_bf + (size_t)row * H_ + kbA;
  const bf16_t* yrow = y_pad + (size_t)row * 32 + kbA;
  const bf16_t* brow[6];
#pragma unroll
  for (int nt = 0; nt < 6; ++nt)
    brow[nt] = attB + (size_t)((nt0 + nt) * 16 + rsel) * 288 + kbB;

  v8f acc[6];
#pragma unroll
  for (int t = 0; t < 6; ++t)
#pragma unroll
    for (int e = 0; e < 8; ++e) acc[t][e] = 0.0f;

#pragma unroll
  for (int kc = 0; kc < 288; kc += 32) {
    v8bf lo = (kc < H_)      ? ldg8(hrow + kc)      : ldg8(yrow + (kc - H_));
    v8bf hi = (kc + 16 < H_) ? ldg8(hrow + kc + 16) : ldg8(yrow + (kc + 16 - H_));
    const v16bf a = cat16(lo, hi);
#pragma unroll
    for (int nt = 0; nt < 6; ++nt) {
      const v16bf bb = cat16(ldg8(brow[nt] + kc), ldg8(brow[nt] + kc + 16));
      acc[nt] = __builtin_amdgcn_wmma_f32_16x16x32_bf16(
          false, a, false, bb, (short)0, acc[nt], false, false);
    }
  }
#pragma unroll
  for (int nt = 0; nt < 6; ++nt) {
#pragma unroll
    for (int v = 0; v < 8; ++v) {
      const int rl = wm * 16 + v + ((lane >> 4) << 3);
      const int c  = (nt0 + nt) * 16 + rsel;
      SC[rl][c] = acc[nt][v] + att_bp[c];
    }
  }
  __syncthreads();
  if (tid < 32) {
    float m = -3.0e38f;
    for (int c = 0; c < S_; ++c) m = fmaxf(m, SC[tid][c]);
    float s = 0.0f;
    for (int c = 0; c < S_; ++c) s += expf(SC[tid][c] - m);
    SM[tid] = m;
    SR[tid] = 1.0f / s;
  }
  __syncthreads();
  for (int idx = tid; idx < 32 * S_; idx += 256) {
    const int r = idx / S_, c = idx % S_;
    attn[(size_t)(m0 + r) * S_ + c] = expf(SC[r][c] - SM[r]) * SR[r];
  }
}

// ---- attention combine: L2-resident enc_out (bf16, 88MB < 192MB L2) ----
__global__ __launch_bounds__(256) void combine_kernel(
    const float* __restrict__ attn, const bf16_t* __restrict__ enc_out,
    bf16_t* __restrict__ comb_bf) {
  const int b = blockIdx.x, tid = threadIdx.x;
  __shared__ float A[S_];
  for (int i = tid; i < S_; i += 256) A[i] = attn[(size_t)b * S_ + i];
  __syncthreads();
  const bf16_t* e = enc_out + (size_t)b * S_ * H_ + tid;
  float acc = 0.0f;
#pragma unroll 4
  for (int s = 0; s < S_; ++s) acc += A[s] * (float)e[(size_t)s * H_];
  comb_bf[(size_t)b * H_ + tid] = (bf16_t)acc;
}

__global__ __launch_bounds__(256) void out_kernel(
    const float* __restrict__ h, const float* __restrict__ out_W,
    const float* __restrict__ out_b, bf16_t* __restrict__ y_pad,
    float* __restrict__ outs, int p) {
  const int idx = blockIdx.x * 256 + threadIdx.x;  // B*F = 8192
  const int b = idx >> 4, f = idx & 15;
  float acc = out_b[f];
  const float* hr = h + (size_t)b * H_;
  const float* wr = out_W + (size_t)f * H_;
#pragma unroll 4
  for (int j = 0; j < H_; ++j) acc += hr[j] * wr[j];
  y_pad[(size_t)b * 32 + f] = (bf16_t)acc;  // upper 16 stay zero from init
  outs[(size_t)b * (P_ * F_) + p * F_ + f] = acc;
}

__global__ __launch_bounds__(256) void final_kernel(
    const float* __restrict__ outs, const float* __restrict__ lin_W,
    const float* __restrict__ lin_b, float* __restrict__ out) {
  const int idx = blockIdx.x * 256 + threadIdx.x;  // B*P = 24576
  if (idx >= B_ * P_) return;
  const int b = idx / P_, p = idx % P_;
  float acc = lin_b[0];
#pragma unroll
  for (int f = 0; f < F_; ++f)
    acc += outs[(size_t)b * (P_ * F_) + p * F_ + f] * lin_W[f];
  out[idx] = acc;
}

// ---- prep: pack gate-major B matrix [4*H][KTOT] = [Whh_g | Wih_g | 0] ----
__global__ __launch_bounds__(256) void build_gruB_kernel(
    const float* __restrict__ Whh, const float* __restrict__ Wih,
    int Kx, int Ktot, bf16_t* __restrict__ Bm) {
  const int idx = blockIdx.x * 256 + threadIdx.x;
  if (idx >= 4 * H_ * Ktot) return;
  const int row = idx / Ktot;  // g*H + col
  const int k   = idx % Ktot;
  const int g   = row >> 8;
  const int col = row & (H_ - 1);
  const int gw  = (g == 1 ? H_ : (g >= 2 ? 2 * H_ : 0)) + col;
  float v = 0.0f;
  if (k < H_) {
    if (g != 3) v = Whh[(size_t)gw * H_ + k];  // g==3 (nx): zero h-region
  } else {
    const int kk = k - H_;
    if (kk < Kx && g != 2) v = Wih[(size_t)gw * Kx + kk];  // g==2 (nh): zero x-region
  }
  Bm[idx] = (bf16_t)v;
}

// attB[NATT][288]: K padded 272->288, N padded 336->384 (zero rows); plus
// padded bias att_bp[NATT]
__global__ __launch_bounds__(256) void build_attB_kernel(
    const float* __restrict__ attW, const float* __restrict__ att_b,
    bf16_t* __restrict__ attB, float* __restrict__ att_bp) {
  const int idx = blockIdx.x * 256 + threadIdx.x;
  if (idx >= NATT * 288) return;
  const int col = idx / 288, k = idx % 288;
  float v = 0.0f;
  if (col < S_ && k < (H_ + F_)) v = attW[(size_t)col * (H_ + F_) + k];
  attB[idx] = (bf16_t)v;
  if (idx < NATT) att_bp[idx] = (idx < S_) ? att_b[idx] : 0.0f;
}

// xpad[t][b][32]: xb padded 16->32, per-step contiguous
__global__ __launch_bounds__(256) void build_xpad_kernel(
    const float* __restrict__ xb, bf16_t* __restrict__ xpad) {
  const int idx = blockIdx.x * 256 + threadIdx.x;
  if (idx >= S_ * B_ * 32) return;
  const int t = idx / (B_ * 32);
  const int r = idx % (B_ * 32);
  const int b = r >> 5, f = r & 31;
  xpad[idx] = (bf16_t)(f < F_ ? xb[(size_t)b * S_ * F_ + (size_t)t * F_ + f] : 0.0f);
}

__global__ __launch_bounds__(256) void init_kernel(const float* __restrict__ xb,
                                                   float* __restrict__ h0f,
                                                   bf16_t* __restrict__ h0b,
                                                   bf16_t* __restrict__ y_pad) {
  const int i = blockIdx.x * 256 + threadIdx.x;
  if (i < B_ * H_) { h0f[i] = 0.0f; h0b[i] = (bf16_t)0.0f; }
  if (i < B_ * 32) {
    const int b = i >> 5, f = i & 31;
    const float v =
        (f < F_) ? xb[(size_t)b * S_ * F_ + (size_t)(S_ - 1) * F_ + f] : 0.0f;
    y_pad[i] = (bf16_t)v;
  }
}

extern "C" void kernel_launch(void* const* d_in, const int* in_sizes, int n_in,
                              void* d_out, int out_size, void* d_ws, size_t ws_size,
                              hipStream_t stream) {
  (void)in_sizes; (void)n_in; (void)out_size; (void)ws_size;
  const float* xb      = (const float*)d_in[0];
  const float* enc_Wih = (const float*)d_in[1];
  const float* enc_Whh = (const float*)d_in[2];
  const float* enc_bih = (const float*)d_in[3];
  const float* enc_bhh = (const float*)d_in[4];
  const float* att_W   = (const float*)d_in[5];
  const float* att_b   = (const float*)d_in[6];
  const float* dec_Wih = (const float*)d_in[7];
  const float* dec_Whh = (const float*)d_in[8];
  const float* dec_bih = (const float*)d_in[9];
  const float* dec_bhh = (const float*)d_in[10];
  const float* out_W   = (const float*)d_in[11];
  const float* out_b   = (const float*)d_in[12];
  const float* lin_W   = (const float*)d_in[13];
  const float* lin_b   = (const float*)d_in[14];

  char* w = (char*)d_ws;
  size_t off = 0;
  auto alloc = [&](size_t bytes) -> char* {
    char* p = w + off;
    off = (off + bytes + 255) & ~(size_t)255;
    return p;
  };
  bf16_t* enc_out_bf = (bf16_t*)alloc((size_t)B_ * S_ * H_ * 2);
  float*  hf[2];
  hf[0] = (float*)alloc((size_t)B_ * H_ * 4);
  hf[1] = (float*)alloc((size_t)B_ * H_ * 4);
  bf16_t* hb[2];
  hb[0] = (bf16_t*)alloc((size_t)B_ * H_ * 2);
  hb[1] = (bf16_t*)alloc((size_t)B_ * H_ * 2);
  bf16_t* xpad   = (bf16_t*)alloc((size_t)S_ * B_ * 32 * 2);
  bf16_t* encB   = (bf16_t*)alloc((size_t)4 * H_ * 288 * 2);
  bf16_t* decB   = (bf16_t*)alloc((size_t)4 * H_ * 512 * 2);
  bf16_t* attB   = (bf16_t*)alloc((size_t)NATT * 288 * 2);
  float*  att_bp = (float*)alloc((size_t)NATT * 4);
  float*  attn_f = (float*)alloc((size_t)B_ * S_ * 4);
  bf16_t* comb   = (bf16_t*)alloc((size_t)B_ * H_ * 2);
  bf16_t* y_pad  = (bf16_t*)alloc((size_t)B_ * 32 * 2);
  float*  outs_f = (float*)alloc((size_t)B_ * P_ * F_ * 4);

  build_gruB_kernel<<<(4 * H_ * 288 + 255) / 256, 256, 0, stream>>>(
      enc_Whh, enc_Wih, F_, 288, encB);
  build_gruB_kernel<<<(4 * H_ * 512 + 255) / 256, 256, 0, stream>>>(
      dec_Whh, dec_Wih, H_, 512, decB);
  build_attB_kernel<<<(NATT * 288 + 255) / 256, 256, 0, stream>>>(
      att_W, att_b, attB, att_bp);
  build_xpad_kernel<<<(S_ * B_ * 32 + 255) / 256, 256, 0, stream>>>(xb, xpad);
  init_kernel<<<(B_ * H_ + 255) / 256, 256, 0, stream>>>(xb, hf[0], hb[0], y_pad);

  dim3 ggrid(B_ / 64, H_ / 32);
  // encoder scan: h double-buffered across launches (no intra-launch races)
  for (int t = 0; t < S_; ++t) {
    gru_step_kernel<32, 288, true><<<ggrid, 256, 0, stream>>>(
        hb[t & 1], hf[t & 1], xpad + (size_t)t * B_ * 32, encB,
        enc_bih, enc_bhh, hf[(t + 1) & 1], hb[(t + 1) & 1],
        enc_out_bf + (size_t)t * H_);
  }
  // decoder scan
  for (int p = 0; p < P_; ++p) {
    att_kernel<<<B_ / 32, 256, 0, stream>>>(hb[p & 1], y_pad, attB, att_bp, attn_f);
    combine_kernel<<<B_, 256, 0, stream>>>(attn_f, enc_out_bf, comb);
    gru_step_kernel<256, 512, false><<<ggrid, 256, 0, stream>>>(
        hb[p & 1], hf[p & 1], comb, decB,
        dec_bih, dec_bhh, hf[(p + 1) & 1], hb[(p + 1) & 1], (bf16_t*)nullptr);
    out_kernel<<<B_ * F_ / 256, 256, 0, stream>>>(hf[(p + 1) & 1], out_W, out_b,
                                                  y_pad, outs_f, p);
  }
  final_kernel<<<(B_ * P_ + 255) / 256, 256, 0, stream>>>(outs_f, lin_W, lin_b,
                                                          (float*)d_out);
}
